// RecurrentApproximator_75737453297800
// MI455X (gfx1250) — compile-verified, matrix operand order checked by reading, not run on hardware
//
#include <hip/hip_runtime.h>

// ---------------------------------------------------------------------------
// RecurrentApproximator forward for MI455X (gfx1250, wave32, WMMA f16->f32)
// B=256, T=256, H=256, L=4, TMAX=256.  Batch-row partitioned persistent RNN:
// 16 workgroups x 16 batch rows, no inter-workgroup sync needed anywhere.
// keys/u (64MB f16) stay resident in the 192MB L2 for the whole decode loop.
// A laundered zero *offset* (not pointer) is added to the weight bases each
// timestep: blocks LICM from hoisting the loop-invariant weight fragment
// loads (which previously caused scratch spills) while preserving pointer
// provenance so the loads lower to global_load_b128 (not flat_load_b128,
// which would tie up DScnt and the LDS path).
// ---------------------------------------------------------------------------

#define Bsz 256
#define Tt  256
#define Hh  256
#define Ll  4
#define TD  255
#define BOSV (-10001.0f)

typedef _Float16 v8h  __attribute__((ext_vector_type(8)));
typedef _Float16 v16h __attribute__((ext_vector_type(16)));
typedef float    v8f  __attribute__((ext_vector_type(8)));

#if defined(__has_builtin)
# if __has_builtin(__builtin_amdgcn_tanhf)
#  define fast_tanh(x) __builtin_amdgcn_tanhf(x)
# elif __has_builtin(__builtin_amdgcn_tanh_f32)
#  define fast_tanh(x) __builtin_amdgcn_tanh_f32(x)
# else
#  define fast_tanh(x) tanhf(x)
# endif
#else
# define fast_tanh(x) tanhf(x)
#endif

// Opaque zero: blocks loop-invariant code motion of address computations and
// loads derived from (base + zoff) without destroying pointer provenance.
#define LAUNDER_OFF(o) asm volatile("" : "+s"(o))

// Load one 16-half WMMA A/B fragment slice for this lane.
// p already points at row_base + khi; halves [0..7] = p[0..7] (K+0..7),
// halves [8..15] = p[16..23] (K+16..23) per the 16-bit A/B VGPR layout.
__device__ inline v16h frag_ld(const _Float16* p) {
  v8h lo = *(const v8h*)(p);
  v8h hi = *(const v8h*)(p + 16);
  v16h f;
#pragma unroll
  for (int i = 0; i < 8; ++i) { f[i] = lo[i]; f[i + 8] = hi[i]; }
  return f;
}

// C(16x16) tile for wave `wv`: A rows from up to 3 LDS/global segments of
// 16x256 halves (pitch 256), B columns = rows of packed weight matrix W
// (wrow = W + j*NSEG*256, j = wv*16 + (lane&15)).  K = NSEG*256.
template <int NSEG>
__device__ inline v8f mm_tile(const _Float16* s0, const _Float16* s1,
                              const _Float16* s2, const _Float16* wrow,
                              int lane) {
  const int khi = (lane >> 4) << 3;   // 0 or 8
  const int m   = lane & 15;          // A row / B column
  v8f acc = {};
  const _Float16* segs[3] = { s0, s1, s2 };
#pragma unroll
  for (int s = 0; s < NSEG; ++s) {
    const _Float16* ab = segs[s] + m * 256 + khi;
    const _Float16* wb = wrow + s * 256 + khi;
#pragma unroll
    for (int kb = 0; kb < 256; kb += 32) {
      v16h a = frag_ld(ab + kb);
      v16h b = frag_ld(wb + kb);
      acc = __builtin_amdgcn_wmma_f32_16x16x32_f16(
          false, a, false, b, (short)0, acc, false, false);
    }
  }
  return acc;
}

// Scatter C tile to a 16x256 f16 row-major LDS block with tanh(acc+bias).
__device__ inline void store_tanh(v8f acc, float bias, _Float16* D, int lane,
                                  int wv) {
  const int n = lane & 15, mh = (lane >> 4) << 3;
#pragma unroll
  for (int r = 0; r < 8; ++r)
    D[(r + mh) * 256 + wv * 16 + n] = (_Float16)fast_tanh(acc[r] + bias);
}

__device__ inline void store_lin(v8f acc, float bias, _Float16* D, int lane,
                                 int wv) {
  const int n = lane & 15, mh = (lane >> 4) << 3;
#pragma unroll
  for (int r = 0; r < 8; ++r)
    D[(r + mh) * 256 + wv * 16 + n] = (_Float16)(acc[r] + bias);
}

// ------------------------- weight repacking -------------------------------
// Encoder: Wcat[l][j][0:256]=Wih[l][j][:], [256:512]=Whh[l][j][:]  (f16)
__global__ void pack_enc_k(const float* __restrict__ Wih,
                           const float* __restrict__ Whh,
                           _Float16* __restrict__ dst) {
  int i = blockIdx.x * 256 + threadIdx.x;            // [l][j][k<512]
  int k = i & 511, j = (i >> 9) & 255, l = i >> 17;
  float v = (k < 256) ? Wih[((l * 256 + j) << 8) + k]
                      : Whh[((l * 256 + j) << 8) + (k - 256)];
  dst[i] = (_Float16)v;
}

// Decoder layer0: [j][0:512]=Wih0[j][:], [512:768]=Whh[0][j][:]
__global__ void pack_dec0_k(const float* __restrict__ Wih0,
                            const float* __restrict__ Whh,
                            _Float16* __restrict__ dst) {
  int i = blockIdx.x * 256 + threadIdx.x;            // 256*768
  int j = i / 768, k = i - j * 768;
  float v = (k < 512) ? Wih0[j * 512 + k] : Whh[(j << 8) + (k - 512)];
  dst[i] = (_Float16)v;
}

// Decoder layers 1..3: [l-1][j][0:256]=Wih[l-1][j], [256:512]=Whh[l][j]
__global__ void pack_dec_k(const float* __restrict__ Wih,
                           const float* __restrict__ Whh,
                           _Float16* __restrict__ dst) {
  int i = blockIdx.x * 256 + threadIdx.x;            // 3*256*512
  int k = i & 511, j = (i >> 9) & 255, lp = i >> 17; // lp = l-1
  float v = (k < 256) ? Wih[((lp * 256 + j) << 8) + k]
                      : Whh[(((lp + 1) * 256 + j) << 8) + (k - 256)];
  dst[i] = (_Float16)v;
}

__global__ void pack_f16_k(const float* __restrict__ src,
                           _Float16* __restrict__ dst) {
  int i = blockIdx.x * 256 + threadIdx.x;
  dst[i] = (_Float16)src[i];
}

__global__ void addv_k(const float* __restrict__ a, const float* __restrict__ b,
                       float* __restrict__ dst) {
  int i = blockIdx.x * 256 + threadIdx.x;
  dst[i] = a[i] + b[i];
}

// ----------------------------- encoder ------------------------------------
// 16 persistent workgroups, 16 batch rows each; loops all 256 timesteps.
__global__ __launch_bounds__(512) void encoder_kernel(
    const float* __restrict__ inputs, const float* __restrict__ emb_w,
    const float* __restrict__ emb_b, const _Float16* __restrict__ Wenc,
    const float* __restrict__ benc, _Float16* __restrict__ keys,
    float* __restrict__ hfinal) {
  __shared__ _Float16 Ebuf[16 * 256];
  __shared__ _Float16 Hst[4][16 * 256];
  const int tid = threadIdx.x, lane = tid & 31, wv = tid >> 5;
  const int bb0 = blockIdx.x * 16;
  const int j = wv * 16 + (lane & 15);

  for (int i = tid; i < 16 * 256; i += 512) {
#pragma unroll
    for (int l = 0; l < 4; ++l) Hst[l][i] = (_Float16)0.f;
  }
  __syncthreads();

  for (int t = 0; t < Tt; ++t) {
    // Opaque zero offset: weight loads stay inside the loop (streamed from
    // L2) instead of hoist-and-spill, but keep global provenance.
    size_t zoff = 0;
    LAUNDER_OFF(zoff);
    const _Float16* Wt = Wenc + zoff;

    for (int i = tid; i < 16 * 256; i += 512) {
      int r = i >> 8, h = i & 255;
      Ebuf[i] = (_Float16)(inputs[(bb0 + r) * Tt + t] * emb_w[h] + emb_b[h]);
    }
    __syncthreads();
#pragma unroll
    for (int l = 0; l < 4; ++l) {
      const _Float16* xseg = (l == 0) ? (const _Float16*)Ebuf
                                      : (const _Float16*)Hst[l - 1];
      v8f acc = mm_tile<2>(xseg, Hst[l], nullptr,
                           Wt + ((size_t)l * 256 + j) * 512, lane);
      float bias = benc[l * 256 + j];
      __syncthreads();                 // all reads of Hst[l] done
      store_tanh(acc, bias, Hst[l], lane, wv);  // in-place h update
      __syncthreads();
    }
    for (int i = tid; i < 16 * 256; i += 512) {
      int r = i >> 8, h = i & 255;
      keys[(((size_t)(bb0 + r)) * Tt + t) * Hh + h] = Hst[3][i];
    }
    __syncthreads();
  }
  for (int i = tid; i < 16 * 256; i += 512) {
    int r = i >> 8, h = i & 255;
#pragma unroll
    for (int l = 0; l < 4; ++l)
      hfinal[(((size_t)l) * Bsz + bb0 + r) * Hh + h] = (float)Hst[l][i];
  }
}

// ---------------------- u = keys @ Ua^T + Ua_b (parallel) ------------------
__global__ __launch_bounds__(512) void ua_kernel(
    const _Float16* __restrict__ keys, const _Float16* __restrict__ Ua,
    const float* __restrict__ Ua_b, _Float16* __restrict__ u) {
  const int lane = threadIdx.x & 31, wv = threadIdx.x >> 5;
  const int j = wv * 16 + (lane & 15);
  const size_t rb = (size_t)blockIdx.x * 16;   // 16 rows of (B*T)
  v8f acc = mm_tile<1>(keys + rb * 256, nullptr, nullptr,
                       Ua + (size_t)j * 256, lane);
  const float bias = Ua_b[j];
  const int n = lane & 15, mh = (lane >> 4) << 3;
#pragma unroll
  for (int r = 0; r < 8; ++r)
    u[(rb + r + mh) * 256 + wv * 16 + n] = (_Float16)(acc[r] + bias);
}

// ----------------------------- decoder ------------------------------------
__global__ __launch_bounds__(512) void decoder_kernel(
    const float* __restrict__ targets, const float* __restrict__ emb_w,
    const float* __restrict__ emb_b, const _Float16* __restrict__ Wdec0,
    const _Float16* __restrict__ Wdec, const float* __restrict__ bdec,
    const _Float16* __restrict__ Wa, const float* __restrict__ Wa_b,
    const _Float16* __restrict__ keys, const _Float16* __restrict__ u,
    const float* __restrict__ Va_w, const float* __restrict__ Va_b,
    const float* __restrict__ head_w, const float* __restrict__ head_b,
    const float* __restrict__ hfinal, float* __restrict__ out) {
  __shared__ _Float16 Ebuf[16 * 256];
  __shared__ _Float16 Cbuf[16 * 256];
  __shared__ _Float16 WS[16 * 256];   // attention w, then scores/weights
  __shared__ _Float16 Hst[4][16 * 256];
  const int tid = threadIdx.x, lane = tid & 31, wv = tid >> 5;
  const int bb0 = blockIdx.x * 16;
  const int j = wv * 16 + (lane & 15);

  for (int i = tid; i < 16 * 256; i += 512) {
    int r = i >> 8, h = i & 255;
#pragma unroll
    for (int l = 0; l < 4; ++l)
      Hst[l][i] = (_Float16)hfinal[(((size_t)l) * Bsz + bb0 + r) * Hh + h];
  }
  if (tid < 16) out[((size_t)(bb0 + tid)) * 256] = BOSV;

  float va[8], hw[8];
#pragma unroll
  for (int i = 0; i < 8; ++i) {
    va[i] = Va_w[lane * 8 + i];
    hw[i] = head_w[lane * 8 + i];
  }
  const float vab = Va_b[0], hb = head_b[0];
  const _Float16* urow = u + (((size_t)(bb0 + wv)) * Tt) * Hh;
  const _Float16* krow = keys + (((size_t)(bb0 + wv)) * Tt) * Hh;
  __syncthreads();

  for (int t = 0; t < TD; ++t) {
    // Opaque zero offset per step (see header comment).
    size_t zoff = 0;
    LAUNDER_OFF(zoff);
    const _Float16* Wa_t = Wa + zoff;
    const _Float16* W0_t = Wdec0 + zoff;
    const _Float16* Wd_t = Wdec + zoff;

    // --- w = q @ Wa^T + Wa_b  (q = Hst[3]) ---
    {
      v8f acc = mm_tile<1>(Hst[3], nullptr, nullptr,
                           Wa_t + (size_t)j * 256, lane);
      store_lin(acc, Wa_b[j], WS, lane, wv);
    }
    __syncthreads();
    // --- scores: wave wv owns batch row wv; lanes split h ---
    float wq[8];
#pragma unroll
    for (int i = 0; i < 8; ++i) wq[i] = (float)WS[wv * 256 + lane * 8 + i];
    __syncthreads();
    for (int tp = 0; tp < Tt; ++tp) {
      v8h uu = *(const v8h*)(urow + (size_t)tp * Hh + lane * 8);
      float s = 0.f;
#pragma unroll
      for (int i = 0; i < 8; ++i) s += va[i] * fast_tanh(wq[i] + (float)uu[i]);
#pragma unroll
      for (int mk = 16; mk > 0; mk >>= 1) s += __shfl_xor(s, mk, 32);
      if (lane == 0) WS[wv * 256 + tp] = (_Float16)(s + vab);
    }
    __syncthreads();
    // --- softmax over T within row wv ---
    {
      float vl[8], mx = -3.0e38f;
#pragma unroll
      for (int i = 0; i < 8; ++i) {
        vl[i] = (float)WS[wv * 256 + lane * 8 + i];
        mx = fmaxf(mx, vl[i]);
      }
#pragma unroll
      for (int mk = 16; mk > 0; mk >>= 1) mx = fmaxf(mx, __shfl_xor(mx, mk, 32));
      float sm = 0.f;
#pragma unroll
      for (int i = 0; i < 8; ++i) { vl[i] = __expf(vl[i] - mx); sm += vl[i]; }
#pragma unroll
      for (int mk = 16; mk > 0; mk >>= 1) sm += __shfl_xor(sm, mk, 32);
      float inv = 1.0f / sm;
#pragma unroll
      for (int i = 0; i < 8; ++i)
        WS[wv * 256 + lane * 8 + i] = (_Float16)(vl[i] * inv);
    }
    __syncthreads();
    // --- ctx[b] = sum_t wts[t] * keys[b,t,:] ---
    {
      float c[8] = {};
      for (int tp = 0; tp < Tt; ++tp) {
        float wt = (float)WS[wv * 256 + tp];
        v8h kk = *(const v8h*)(krow + (size_t)tp * Hh + lane * 8);
#pragma unroll
        for (int i = 0; i < 8; ++i) c[i] += wt * (float)kk[i];
      }
#pragma unroll
      for (int i = 0; i < 8; ++i) Cbuf[wv * 256 + lane * 8 + i] = (_Float16)c[i];
    }
    // --- embedding of teacher-forced input ---
    for (int i = tid; i < 16 * 256; i += 512) {
      int r = i >> 8, h = i & 255;
      float x = BOSV;
      if (t > 0) x = targets[((size_t)(bb0 + r)) * TD + (t - 1)];
      Ebuf[i] = (_Float16)(x * emb_w[h] + emb_b[h]);
    }
    __syncthreads();
    // --- layer 0: A = [e | ctx | h0], K = 768 ---
    {
      v8f acc = mm_tile<3>(Ebuf, Cbuf, Hst[0], W0_t + (size_t)j * 768, lane);
      float bias = bdec[j];
      __syncthreads();
      store_tanh(acc, bias, Hst[0], lane, wv);
      __syncthreads();
    }
    // --- layers 1..3: A = [h_{l-1}(new) | h_l(old)], K = 512 ---
#pragma unroll
    for (int l = 1; l < 4; ++l) {
      v8f acc = mm_tile<2>(Hst[l - 1], Hst[l], nullptr,
                           Wd_t + (((size_t)(l - 1)) * 256 + j) * 512, lane);
      float bias = bdec[l * 256 + j];
      __syncthreads();
      store_tanh(acc, bias, Hst[l], lane, wv);
      __syncthreads();
    }
    // --- head: pred[b] = h3 . head_w + head_b ---
    {
      float s = 0.f;
#pragma unroll
      for (int i = 0; i < 8; ++i)
        s += (float)Hst[3][wv * 256 + lane * 8 + i] * hw[i];
#pragma unroll
      for (int mk = 16; mk > 0; mk >>= 1) s += __shfl_xor(s, mk, 32);
      if (lane == 0) out[((size_t)(bb0 + wv)) * 256 + t + 1] = s + hb;
    }
    __syncthreads();
  }
}

// ----------------------------- launch -------------------------------------
extern "C" void kernel_launch(void* const* d_in, const int* in_sizes, int n_in,
                              void* d_out, int out_size, void* d_ws,
                              size_t ws_size, hipStream_t stream) {
  (void)in_sizes; (void)n_in; (void)out_size; (void)ws_size;
  const float* inputs   = (const float*)d_in[0];
  const float* targets  = (const float*)d_in[1];
  const float* emb_w    = (const float*)d_in[2];
  const float* emb_b    = (const float*)d_in[3];
  const float* enc_Wih  = (const float*)d_in[4];
  const float* enc_Whh  = (const float*)d_in[5];
  const float* enc_bih  = (const float*)d_in[6];
  const float* enc_bhh  = (const float*)d_in[7];
  const float* dec_Wih0 = (const float*)d_in[8];
  const float* dec_Wih  = (const float*)d_in[9];
  const float* dec_Whh  = (const float*)d_in[10];
  const float* dec_bih  = (const float*)d_in[11];
  const float* dec_bhh  = (const float*)d_in[12];
  const float* Wa_w     = (const float*)d_in[13];
  const float* Wa_b     = (const float*)d_in[14];
  const float* Ua_w     = (const float*)d_in[15];
  const float* Ua_b     = (const float*)d_in[16];
  const float* Va_w     = (const float*)d_in[17];
  const float* Va_b     = (const float*)d_in[18];
  const float* head_w   = (const float*)d_in[19];
  const float* head_b   = (const float*)d_in[20];
  float* out = (float*)d_out;

  char* ws = (char*)d_ws;
  constexpr size_t SZ_WENC  = (size_t)Ll * Hh * 2 * Hh * 2;  // 1 MB
  constexpr size_t SZ_WDEC0 = (size_t)Hh * 3 * Hh * 2;       // 384 KB
  constexpr size_t SZ_WDEC  = (size_t)3 * Hh * 2 * Hh * 2;   // 768 KB
  constexpr size_t SZ_SQ    = (size_t)Hh * Hh * 2;           // 128 KB
  constexpr size_t SZ_BIAS  = (size_t)Ll * Hh * 4;           // 4 KB
  constexpr size_t SZ_HFIN  = (size_t)Ll * Bsz * Hh * 4;     // 1 MB
  constexpr size_t SZ_BTH   = (size_t)Bsz * Tt * Hh * 2;     // 32 MB

  size_t o = 0;
  _Float16* Wenc  = (_Float16*)(ws + o); o += SZ_WENC;
  _Float16* Wdec0 = (_Float16*)(ws + o); o += SZ_WDEC0;
  _Float16* Wdec  = (_Float16*)(ws + o); o += SZ_WDEC;
  _Float16* Wa    = (_Float16*)(ws + o); o += SZ_SQ;
  _Float16* Ua    = (_Float16*)(ws + o); o += SZ_SQ;
  float*    benc  = (float*)(ws + o);    o += SZ_BIAS;
  float*    bdec  = (float*)(ws + o);    o += SZ_BIAS;
  float*    hfin  = (float*)(ws + o);    o += SZ_HFIN;
  _Float16* keys  = (_Float16*)(ws + o); o += SZ_BTH;
  _Float16* u     = (_Float16*)(ws + o); o += SZ_BTH;

  pack_enc_k<<<2048, 256, 0, stream>>>(enc_Wih, enc_Whh, Wenc);
  pack_dec0_k<<<768, 256, 0, stream>>>(dec_Wih0, dec_Whh, Wdec0);
  pack_dec_k<<<1536, 256, 0, stream>>>(dec_Wih, dec_Whh, Wdec);
  pack_f16_k<<<256, 256, 0, stream>>>(Wa_w, Wa);
  pack_f16_k<<<256, 256, 0, stream>>>(Ua_w, Ua);
  addv_k<<<4, 256, 0, stream>>>(enc_bih, enc_bhh, benc);
  addv_k<<<4, 256, 0, stream>>>(dec_bih, dec_bhh, bdec);

  encoder_kernel<<<Bsz / 16, 512, 0, stream>>>(inputs, emb_w, emb_b, Wenc,
                                               benc, keys, hfin);
  ua_kernel<<<(Bsz * Tt) / 16, 512, 0, stream>>>(keys, Ua, Ua_b, u);
  decoder_kernel<<<Bsz / 16, 512, 0, stream>>>(
      targets, emb_w, emb_b, Wdec0, Wdec, bdec, Wa, Wa_b, keys, u, Va_w, Va_b,
      head_w, head_b, hfin, out);
}
